// GRU_41901700940187
// MI455X (gfx1250) — matrix operand order repaired and checked by reading, not compiled, not run on hardware
//
#include <hip/hip_runtime.h>
#include <math.h>

// ---------------- CDNA5 WMMA types ----------------
typedef __attribute__((ext_vector_type(16))) __bf16 v16bf;
typedef __attribute__((ext_vector_type(8)))  float  v8f;

// ---------------- problem constants ----------------
#define B_TOT   32768
#define LSEQ    128
#define HDIM    256
#define MV      16          // vocab
#define G3H     768         // 3*H
#define TB      32          // batch rows per workgroup
#define NTHREADS 256        // 8 waves
#define NCHUNK  12          // W_hh streamed in 12 chunks of 64 rows
#define CHUNK_ROWS 64
#define CHUNK_ELEMS (CHUNK_ROWS*HDIM)     // 16384 bf16 = 32 KB
#define CHUNK_UNITS ((CHUNK_ELEMS*2)/8)   // 4096 x 8-byte units per TDM tile

// ---------------- TDM availability ----------------
#if defined(__HIP_DEVICE_COMPILE__) && __has_builtin(__builtin_amdgcn_tensor_load_to_lds) && __has_builtin(__builtin_amdgcn_s_wait_tensorcnt)
#define USE_TDM 1
#else
#define USE_TDM 0
#endif

typedef __attribute__((ext_vector_type(4))) unsigned int u32x4;
typedef __attribute__((ext_vector_type(8))) int          i32x8;
typedef __attribute__((ext_vector_type(4))) int          i32x4;

#if USE_TDM
// LDS byte offset of a generic pointer into workgroup LDS (addrspacecast + ptrtoint)
__device__ __forceinline__ unsigned lds_byte_off(const void* p) {
    return (unsigned)(unsigned long long)
           (__attribute__((address_space(3))) const char*)(const char*)p;
}

// Issue one Tensor-Data-Mover DMA: 32 KB contiguous (1-D tile, 8-byte units)
// from global `gsrc` into LDS at byte offset `ldsoff`.  D# per ISA 8.3/8.4.
__device__ __forceinline__ void tdm_load_chunk(const void* gsrc, unsigned ldsoff) {
    unsigned long long ga = (unsigned long long)(uintptr_t)gsrc;
    u32x4 g0;
    g0[0] = 1u;                                            // count=1 (valid), user mode
    g0[1] = ldsoff;                                        // lds_addr (bytes)
    g0[2] = (unsigned)(ga & 0xFFFFFFFFu);                  // global_addr[31:0]
    g0[3] = (unsigned)((ga >> 32) & 0x01FFFFFFu)           // global_addr[56:32]
            | (2u << 30);                                  // type = 2 ("image")
    i32x8 g1;
    g1[0] = (3 << 16);                                     // wg_mask=0, data_size=3 (8B)
    g1[1] = (int)((CHUNK_UNITS & 0xFFFF) << 16);           // tensor_dim0[15:0]
    g1[2] = (int)(((CHUNK_UNITS >> 16) & 0xFFFF))          // tensor_dim0[31:16]
            | (1 << 16);                                   // tensor_dim1 = 1
    g1[3] = (int)(CHUNK_UNITS << 16);                      // tile_dim0 = 4096 units
    g1[4] = 0;                                             // tile_dim1=0, tile_dim2=0
    g1[5] = (int)CHUNK_UNITS;                              // tensor_dim0_stride[31:0]
    g1[6] = 0;                                             // stride hi / dim1_stride lo
    g1[7] = 0;                                             // dim1_stride hi
    i32x4 z4 = {0, 0, 0, 0};
#if defined(__clang_major__) && (__clang_major__ >= 23)
    i32x8 z8 = {0, 0, 0, 0, 0, 0, 0, 0};
    __builtin_amdgcn_tensor_load_to_lds(g0, g1, z4, z4, z8, 0);
#else
    __builtin_amdgcn_tensor_load_to_lds(g0, g1, z4, z4, 0);
#endif
}
#endif // USE_TDM

// ---------------- prep: fp32 -> bf16 weight images in workspace ----------------
__global__ void gru_prep_bf16(const float* __restrict__ Whh,
                              const float* __restrict__ Wfc,
                              __bf16* __restrict__ wsWhh,
                              __bf16* __restrict__ wsWfc) {
    int i = blockIdx.x * blockDim.x + threadIdx.x;
    if (i < G3H * HDIM) wsWhh[i] = (__bf16)Whh[i];
    if (i < MV * HDIM)  wsWfc[i] = (__bf16)Wfc[i];
}

// ---------------- main persistent-tile GRU kernel ----------------
__global__ __launch_bounds__(NTHREADS)
void gru_wmma_kernel(const int*   __restrict__ x,
                     const float* __restrict__ W_ih,
                     const float* __restrict__ b_ih,
                     const float* __restrict__ b_hh,
                     const float* __restrict__ b_fc,
                     const __bf16* __restrict__ Whh_bf,   // [768][256] row-major
                     const __bf16* __restrict__ Wfc_bf,   // [16][256]  row-major
                     float* __restrict__ out) {
    // LDS plan (~196 KB total, <= 320 KB/WGP)
    __shared__ __align__(128) __bf16 sW[2][CHUNK_ELEMS];  // 64 KB  dbuf W_hh chunk (TDM dest)
    __shared__ float  sGh[TB * G3H];        // 96 KB  pre-activations h@Whh^T
    __shared__ __bf16 sH[TB * HDIM];        // 16 KB  h as WMMA A-operand (bf16)
    __shared__ __bf16 sWfc[MV * HDIM];      //  8 KB  FC weights (bf16)
    __shared__ float  sWih[G3H];            //  3 KB
    __shared__ float  sBih[G3H];            //  3 KB
    __shared__ float  sBhh[G3H];            //  3 KB
    __shared__ float  sLogit[TB * MV];      //  2 KB
    __shared__ float  sXf[TB];
    __shared__ float  sBfc[MV];

    const int tid   = threadIdx.x;
    const int wave  = tid >> 5;
    const int lane  = tid & 31;
    const int l15   = lane & 15;
    const int lhalf = lane >> 4;            // 0: lanes 0-15, 1: lanes 16-31
    const int baseB = blockIdx.x * TB;

    // one-time staging of small params
    for (int i = tid; i < G3H; i += NTHREADS) {
        sWih[i] = W_ih[i]; sBih[i] = b_ih[i]; sBhh[i] = b_hh[i];
    }
    for (int i = tid; i < MV * HDIM; i += NTHREADS) sWfc[i] = Wfc_bf[i];
    if (tid < MV) sBfc[tid] = b_fc[tid];
    for (int i = tid; i < TB * HDIM; i += NTHREADS) sH[i] = (__bf16)0.f;

    // persistent per-thread state: thread j owns h[:, j] (fp32); the log-prob
    // accumulator lives in threads 0..31 (one per batch row)
    float hreg[TB];
#pragma unroll
    for (int r = 0; r < TB; ++r) hreg[r] = 0.f;
    float lp = 0.f;

    const int mtile = wave & 1;   // 2 M-tiles (32 batch rows)
    const int ngrp  = wave >> 1;  // 4 N-tile groups per chunk

#if USE_TDM
    const unsigned ldsW0 = lds_byte_off(&sW[0][0]);
    const unsigned ldsW1 = lds_byte_off(&sW[1][0]);
#endif

    union BU { v16bf v; float4 f[2]; };

    __syncthreads();

    for (int t = 0; t < LSEQ; ++t) {
#if USE_TDM
        // kick the DMA for chunk 0 of this step (buffer 0 is free: its last
        // readers finished before the barriers that ended the previous step)
        if (wave == 0)
            tdm_load_chunk(Whh_bf, ldsW0);
#endif
        // stage scalar input feature x_in[:, t] = (t==0) ? 0 : x[:, t-1]
        if (tid < TB)
            sXf[tid] = (t == 0) ? 0.f : (float)x[(baseB + tid) * LSEQ + (t - 1)];

        // ---- load A fragments (h, bf16) for this wave's M-tile ----
        // ISA A-layout (16x32 bf16): lane<16 -> M=lane, K = {c..c+7, c+16..c+23}
        v16bf afrag[8];
        {
            const __bf16* hrow = &sH[(mtile * 16 + l15) * HDIM];
#pragma unroll
            for (int k = 0; k < 8; ++k) {
                const int cb = k * 32 + (lhalf ? 8 : 0);
                BU u;
                u.f[0] = *(const float4*)(hrow + cb);
                u.f[1] = *(const float4*)(hrow + cb + 16);
                afrag[k] = u.v;
            }
        }

#if !USE_TDM
        // fallback prologue: cooperative copy of W_hh chunk 0 into LDS buffer 0
#pragma unroll
        for (int i = 0; i < 8; ++i) {
            float4 v = *(const float4*)(Whh_bf + (i * NTHREADS + tid) * 8);
            *(float4*)(&sW[0][(i * NTHREADS + tid) * 8]) = v;
        }
#endif

        // ---- GEMM: gh = h @ W_hh^T, streamed over 12 double-buffered chunks ----
        for (int c = 0; c < NCHUNK; ++c) {
#if USE_TDM
            if (wave == 0)
                __builtin_amdgcn_s_wait_tensorcnt(0);   // chunk c landed in LDS
            __syncthreads();                            // visible to all waves
            // start DMA for chunk c+1 into the other buffer; every wave is now
            // past its reads of that buffer (they ended before this barrier)
            if (wave == 0 && (c + 1) < NCHUNK)
                tdm_load_chunk(Whh_bf + (c + 1) * CHUNK_ELEMS,
                               ((c + 1) & 1) ? ldsW1 : ldsW0);
#else
            float4 pf[8];
            if (c + 1 < NCHUNK) {
                const __bf16* src = Whh_bf + (c + 1) * CHUNK_ELEMS;
#pragma unroll
                for (int i = 0; i < 8; ++i)
                    pf[i] = *(const float4*)(src + (i * NTHREADS + tid) * 8);
            }
            if (c + 2 < NCHUNK)
                __builtin_prefetch(Whh_bf + (c + 2) * CHUNK_ELEMS + tid * 64, 0, 0);
            __syncthreads();      // sW[c&1] fully written by all threads
#endif
            // this wave's output tile: rows mtile*16.., cols (c*4+ngrp)*16..
            // B fragments are double-buffered so the two ds_load_b128 for
            // fragment k+1 fly underneath WMMA k (wait dscnt<=2, not 0).
            v8f acc = {};
            const __bf16* wb = &sW[c & 1][(ngrp * 16 + l15) * HDIM];
            BU ub[2];
            {
                const int cb0 = (lhalf ? 8 : 0);
                ub[0].f[0] = *(const float4*)(wb + cb0);
                ub[0].f[1] = *(const float4*)(wb + cb0 + 16);
            }
#pragma unroll
            for (int k = 0; k < 8; ++k) {
                if (k + 1 < 8) {
                    // ISA B-layout (32x16 bf16): lane -> N=lane%16, K = {c..c+7, c+16..c+23}
                    const int cb = (k + 1) * 32 + (lhalf ? 8 : 0);
                    ub[(k + 1) & 1].f[0] = *(const float4*)(wb + cb);
                    ub[(k + 1) & 1].f[1] = *(const float4*)(wb + cb + 16);
                }
                acc = __builtin_amdgcn_wmma_f32_16x16x32_bf16(
                          false, afrag[k], false, ub[k & 1].v, (short)0, acc, false, false);
            }
            // C-layout (16x16 f32): VGPR r -> row r (lanes 0-15) / r+8 (16-31)
            {
                const int colg = (c * 4 + ngrp) * 16 + l15;
                const int rowb = mtile * 16 + lhalf * 8;
#pragma unroll
                for (int r = 0; r < 8; ++r)
                    sGh[(rowb + r) * G3H + colg] = acc[r];
            }
#if !USE_TDM
            if (c + 1 < NCHUNK) {
#pragma unroll
                for (int i = 0; i < 8; ++i)
                    *(float4*)(&sW[(c + 1) & 1][(i * NTHREADS + tid) * 8]) = pf[i];
            }
#endif
        }
        __syncthreads();   // gh and sXf complete

        // ---- gating (fp32): thread j owns column j for all 32 rows ----
        {
            const int j = tid;
            const float wr = sWih[j],            bri = sBih[j],            brh = sBhh[j];
            const float wz = sWih[HDIM + j],     bzi = sBih[HDIM + j],     bzh = sBhh[HDIM + j];
            const float wn = sWih[2 * HDIM + j], bni = sBih[2 * HDIM + j], bnh = sBhh[2 * HDIM + j];
#pragma unroll 4
            for (int r = 0; r < TB; ++r) {
                const float xv  = sXf[r];
                const float ghr = sGh[r * G3H + j];
                const float ghz = sGh[r * G3H + HDIM + j];
                const float ghn = sGh[r * G3H + 2 * HDIM + j];
                const float rg = 1.f / (1.f + __expf(-(xv * wr + bri + ghr + brh)));
                const float zg = 1.f / (1.f + __expf(-(xv * wz + bzi + ghz + bzh)));
                const float pn = xv * wn + bni + rg * (ghn + bnh);
                const float e2 = __expf(-2.f * pn);
                const float ng = (1.f - e2) / (1.f + e2);          // tanh
                const float hn = (1.f - zg) * ng + zg * hreg[r];
                hreg[r] = hn;
                sH[r * HDIM + j] = (__bf16)hn;                      // next A-operand
            }
        }
        __syncthreads();   // new h visible

        // ---- FC logits = h_new @ W_fc^T  (waves 0,1; one 16x16 N-tile) ----
        if (wave < 2) {
            v8f acc = {};
            const __bf16* hrow = &sH[(wave * 16 + l15) * HDIM];
            const __bf16* wrow = &sWfc[l15 * HDIM];
            BU ua[2], ub[2];
            {
                const int cba0 = (lhalf ? 8 : 0);
                const int cbb0 = (lhalf ? 8 : 0);
                ua[0].f[0] = *(const float4*)(hrow + cba0);
                ua[0].f[1] = *(const float4*)(hrow + cba0 + 16);
                ub[0].f[0] = *(const float4*)(wrow + cbb0);
                ub[0].f[1] = *(const float4*)(wrow + cbb0 + 16);
            }
#pragma unroll
            for (int k = 0; k < 8; ++k) {
                if (k + 1 < 8) {
                    const int cba = (k + 1) * 32 + (lhalf ? 8 : 0);
                    const int cbb = (k + 1) * 32 + (lhalf ? 8 : 0);
                    ua[(k + 1) & 1].f[0] = *(const float4*)(hrow + cba);
                    ua[(k + 1) & 1].f[1] = *(const float4*)(hrow + cba + 16);
                    ub[(k + 1) & 1].f[0] = *(const float4*)(wrow + cbb);
                    ub[(k + 1) & 1].f[1] = *(const float4*)(wrow + cbb + 16);
                }
                acc = __builtin_amdgcn_wmma_f32_16x16x32_bf16(
                          false, ua[k & 1].v, false, ub[k & 1].v, (short)0, acc, false, false);
            }
            const int rowb = wave * 16 + lhalf * 8;
#pragma unroll
            for (int r = 0; r < 8; ++r)
                sLogit[(rowb + r) * MV + l15] = acc[r];
        }
        __syncthreads();

        // ---- log-softmax over 16 + gather at target: one thread per row ----
        if (tid < TB) {
            const int tg = x[(baseB + tid) * LSEQ + t];
            float lg[MV], mx = -3.4e38f;
#pragma unroll
            for (int j = 0; j < MV; ++j) {
                lg[j] = sLogit[tid * MV + j] + sBfc[j];
                mx = fmaxf(mx, lg[j]);
            }
            float s = 0.f;
#pragma unroll
            for (int j = 0; j < MV; ++j) s += __expf(lg[j] - mx);
            lp += lg[tg] - mx - __logf(s);
        }
        __syncthreads();
    }

    if (tid < TB) out[baseB + tid] = lp;
}

// ---------------- host launcher ----------------
extern "C" void kernel_launch(void* const* d_in, const int* in_sizes, int n_in,
                              void* d_out, int out_size, void* d_ws, size_t ws_size,
                              hipStream_t stream) {
    const int*   x    = (const int*)  d_in[0];
    const float* W_ih = (const float*)d_in[1];
    const float* W_hh = (const float*)d_in[2];
    const float* b_ih = (const float*)d_in[3];
    const float* b_hh = (const float*)d_in[4];
    const float* W_fc = (const float*)d_in[5];
    const float* b_fc = (const float*)d_in[6];
    float* out = (float*)d_out;

    __bf16* wsWhh = (__bf16*)d_ws;                 // 768*256 bf16
    __bf16* wsWfc = wsWhh + G3H * HDIM;            // 16*256 bf16

    gru_prep_bf16<<<(G3H * HDIM + 255) / 256, 256, 0, stream>>>(W_hh, W_fc, wsWhh, wsWfc);
    gru_wmma_kernel<<<B_TOT / TB, NTHREADS, 0, stream>>>(
        x, W_ih, b_ih, b_hh, b_fc, wsWhh, wsWfc, out);
}
